// SOSLoRALayer_42717744726342
// MI455X (gfx1250) — matrix operand order repaired and checked by reading, not compiled, not run on hardware
//
#include <hip/hip_runtime.h>
#include <stdint.h>

// Problem constants (from reference)
#define M_TOK   4096   // B*S
#define IN_DIM  4096
#define OUT_DIM 4096
#define NEXP    8
#define RANK    8
#define JR      64     // NEXP*RANK: LoRA "extra K" width
#define GATE_T  2.0f

#define KTILES  (IN_DIM / 32)       // 128 K-tiles over W
#define NTILES  (KTILES + JR / 32)  // +2 LoRA K-tiles = 130

typedef __attribute__((ext_vector_type(16))) __bf16 bf16x16;
typedef __attribute__((ext_vector_type(8)))  __bf16 bf16x8;
typedef __attribute__((ext_vector_type(4)))  __bf16 bf16x4;
typedef __attribute__((ext_vector_type(8)))  float  f32x8;
typedef __attribute__((ext_vector_type(4)))  float  f32x4;

// LDS row stride for a 32-wide bf16 K-tile row: 40 bf16 = 80 bytes.
// 80 is a multiple of 16 -> every row base is 16B-aligned for b128 LDS loads,
// and the 20-DWORD bank stride spreads 16 consecutive rows across banks.
#define LDS_STRIDE 40

// Stage a TROWS x 32 fp32 tile (row stride ld) into LDS as bf16.
// Constant trip count: fully unrolled, all global b128 loads issued before
// any conversion/store so ITER loads are in flight simultaneously.
template<int TROWS, int NTHREADS>
__device__ __forceinline__ void stage_tile(const float* __restrict__ src, int ld,
                                           __bf16* __restrict__ dst, int tid) {
  constexpr int ITER = (TROWS * 8) / NTHREADS;
  f32x4 v[ITER];
#pragma unroll
  for (int p = 0; p < ITER; ++p) {
    const int i = tid + p * NTHREADS;
    v[p] = *(const f32x4*)(src + (size_t)(i >> 3) * ld + ((i & 7) << 2));
  }
#pragma unroll
  for (int p = 0; p < ITER; ++p) {
    const int i = tid + p * NTHREADS;
    bf16x4 o;
    o[0] = (__bf16)v[p][0]; o[1] = (__bf16)v[p][1];
    o[2] = (__bf16)v[p][2]; o[3] = (__bf16)v[p][3];
    *(bf16x4*)(dst + (i >> 3) * LDS_STRIDE + ((i & 7) << 2)) = o;
  }
}

// A fragment (16x32 bf16, ISA 7.12.2): lane half h holds row (lane&15),
// K = {h*8..h*8+7} then {16+h*8..16+h*8+7}  -> two 16B LDS loads.
__device__ __forceinline__ bf16x16 load_afrag(const __bf16* lds, int subtile, int lane) {
  const int half = lane >> 4;
  const int row  = subtile * 16 + (lane & 15);
  const __bf16* p = lds + row * LDS_STRIDE + half * 8;
  bf16x8 lo = *(const bf16x8*)(p);
  bf16x8 hi = *(const bf16x8*)(p + 16);
  bf16x16 a;
#pragma unroll
  for (int i = 0; i < 8; ++i) { a[i] = lo[i]; a[i + 8] = hi[i]; }
  return a;
}

// B fragment (32x16 bf16): lane half h holds column (lane&15),
// K = {h*8..h*8+7} then {16+h*8..16+h*8+7}  -> two 16B LDS loads.
// Our B tiles are rows of W (or Bsc): column n of B == row n, K contiguous.
__device__ __forceinline__ bf16x16 load_bfrag(const __bf16* lds, int subtile, int lane) {
  const int half = lane >> 4;
  const int row  = subtile * 16 + (lane & 15);
  const __bf16* p = lds + row * LDS_STRIDE + half * 8;
  bf16x8 lo = *(const bf16x8*)(p);
  bf16x8 hi = *(const bf16x8*)(p + 16);
  bf16x16 b;
#pragma unroll
  for (int i = 0; i < 8; ++i) { b[i] = lo[i]; b[i + 8] = hi[i]; }
  return b;
}

// ---------------------------------------------------------------------------
// Kernel 1: fold gate softmax + expert scales into Bsc[o][j], j = n*RANK + r.
// Bsc is row-major along j, mirroring W's row-major-along-K layout.
// ---------------------------------------------------------------------------
__global__ __launch_bounds__(256) void
sos_lora_prep(const float* __restrict__ loraB, const float* __restrict__ escale,
              const float* __restrict__ glog, float* __restrict__ Bsc) {
  float l[NEXP];
  float mx = -1e30f;
#pragma unroll
  for (int i = 0; i < NEXP; ++i) { l[i] = glog[i] * (1.0f / GATE_T); mx = fmaxf(mx, l[i]); }
  float s = 0.f;
#pragma unroll
  for (int i = 0; i < NEXP; ++i) { l[i] = __expf(l[i] - mx); s += l[i]; }
  const int idx = blockIdx.x * 256 + threadIdx.x;   // idx = o*JR + j
  const int o = idx >> 6, j = idx & 63, n = j >> 3, r = j & 7;
  const float g = l[n] * ((float)NEXP / s);         // softmax * K
  Bsc[idx] = loraB[((size_t)n * OUT_DIM + o) * RANK + r]
           * escale[n * OUT_DIM + o] * g;
}

// ---------------------------------------------------------------------------
// Kernel 2: H[m][j] = sum_i x[m,i] * lora_A[j,i]   (lora_A is [64][4096] flat)
// 64x64x32 tile per block, double-buffered LDS, 8 waves, 2 WMMAs/wave/step.
// ---------------------------------------------------------------------------
__global__ __launch_bounds__(256) void
sos_lora_h(const float* __restrict__ x, const float* __restrict__ loraA,
           float* __restrict__ Hws) {
  __shared__ __bf16 lA[2][64 * LDS_STRIDE];
  __shared__ __bf16 lB[2][64 * LDS_STRIDE];
  const int tid = threadIdx.x, lane = tid & 31, wave = tid >> 5;
  const int wm  = wave & 3;           // m-subtile 0..3
  const int wnb = (wave >> 2) * 2;    // n-subtiles {wnb, wnb+1}
  const int mbase = blockIdx.x * 64;

  f32x8 acc[2] = {};
  const float* aP = x + (size_t)mbase * IN_DIM;

  stage_tile<64, 256>(aP, IN_DIM, lA[0], tid);
  stage_tile<64, 256>(loraA, IN_DIM, lB[0], tid);
  __syncthreads();

  for (int kt = 0; kt < KTILES; ++kt) {
    const int cur = kt & 1;
    if (kt + 1 < KTILES) {
      stage_tile<64, 256>(aP + (kt + 1) * 32, IN_DIM, lA[cur ^ 1], tid);
      stage_tile<64, 256>(loraA + (kt + 1) * 32, IN_DIM, lB[cur ^ 1], tid);
    }
    bf16x16 af = load_afrag(lA[cur], wm, lane);
#pragma unroll
    for (int j = 0; j < 2; ++j) {
      bf16x16 bq = load_bfrag(lB[cur], wnb + j, lane);
      acc[j] = __builtin_amdgcn_wmma_f32_16x16x32_bf16(
          false, af, false, bq, (short)0, acc[j], false, false);
    }
    __syncthreads();
  }
  const int half = lane >> 4, c = lane & 15;
  const int mrow = mbase + wm * 16 + half * 8;
#pragma unroll
  for (int j = 0; j < 2; ++j) {
    const int col = (wnb + j) * 16 + c;
#pragma unroll
    for (int v = 0; v < 8; ++v)
      Hws[(size_t)(mrow + v) * JR + col] = acc[j][v];
  }
}

// ---------------------------------------------------------------------------
// Kernel 3: main fused GEMM. C = bf16(x)*bf16(W)^T  (K=4096)
//                            + bf16(H)*bf16(Bsc)^T (K=64, the LoRA delta)
//                            + bias
// 128x128x32 tile, double-buffered LDS (one barrier per K-step),
// 8 waves, each wave owns 4x2 16x16 f32 accumulators (8 WMMAs per step).
// ---------------------------------------------------------------------------
__global__ __launch_bounds__(256) void
sos_lora_main(const float* __restrict__ x, const float* __restrict__ W,
              const float* __restrict__ bias, const float* __restrict__ Hws,
              const float* __restrict__ Bsc, float* __restrict__ out) {
  __shared__ __bf16 lA[2][128 * LDS_STRIDE];
  __shared__ __bf16 lB[2][128 * LDS_STRIDE];
  const int tid = threadIdx.x, lane = tid & 31, wave = tid >> 5;
  const int wm = wave >> 2;   // 0..1 : m-subtiles wm*4 .. wm*4+3
  const int wn = wave & 3;    // 0..3 : n-subtiles wn*2 .. wn*2+1
  const int mbase = blockIdx.x * 128;
  const int nbase = blockIdx.y * 128;

  f32x8 acc[4][2] = {};

  const float* aP = x + (size_t)mbase * IN_DIM;
  const float* bP = W + (size_t)nbase * IN_DIM;
  const float* hP = Hws + (size_t)mbase * JR;
  const float* sP = Bsc + (size_t)nbase * JR;

  // Stage K-tile t into LDS buffer buf: tiles [0,128) from x/W,
  // tiles 128..129 are the LoRA "extra K" from H/Bsc (same layout, ld=JR).
  auto stage_t = [&](int t, int buf) {
    if (t < KTILES) {
      stage_tile<128, 256>(aP + t * 32, IN_DIM, lA[buf], tid);
      stage_tile<128, 256>(bP + t * 32, IN_DIM, lB[buf], tid);
    } else {
      const int kk = (t - KTILES) * 32;
      stage_tile<128, 256>(hP + kk, JR, lA[buf], tid);
      stage_tile<128, 256>(sP + kk, JR, lB[buf], tid);
    }
  };

  auto mma_step = [&](int buf) {
    bf16x16 af[4], bq[2];
#pragma unroll
    for (int i = 0; i < 4; ++i) af[i] = load_afrag(lA[buf], wm * 4 + i, lane);
#pragma unroll
    for (int j = 0; j < 2; ++j) bq[j] = load_bfrag(lB[buf], wn * 2 + j, lane);
#pragma unroll
    for (int i = 0; i < 4; ++i)
#pragma unroll
      for (int j = 0; j < 2; ++j)
        acc[i][j] = __builtin_amdgcn_wmma_f32_16x16x32_bf16(
            false, af[i], false, bq[j], (short)0, acc[i][j], false, false);
  };

  stage_t(0, 0);
  __syncthreads();
  for (int t = 0; t < NTILES; ++t) {
    if (t + 1 < NTILES) stage_t(t + 1, (t + 1) & 1);   // overlap with WMMAs
    mma_step(t & 1);
    __syncthreads();
  }

  // Epilogue: C layout (ISA 7.12.2): VGPR v, lane l -> row v + 8*(l>>4), col l&15.
  const int half = lane >> 4, c = lane & 15;
#pragma unroll
  for (int i = 0; i < 4; ++i) {
    const int mrow = mbase + (wm * 4 + i) * 16 + half * 8;
#pragma unroll
    for (int j = 0; j < 2; ++j) {
      const int col = nbase + (wn * 2 + j) * 16 + c;
      const float bv = bias[col];
#pragma unroll
      for (int v = 0; v < 8; ++v)
        out[(size_t)(mrow + v) * OUT_DIM + col] = acc[i][j][v] + bv;
    }
  }
}

extern "C" void kernel_launch(void* const* d_in, const int* in_sizes, int n_in,
                              void* d_out, int out_size, void* d_ws, size_t ws_size,
                              hipStream_t stream) {
  (void)in_sizes; (void)n_in; (void)out_size; (void)ws_size;
  const float* x  = (const float*)d_in[0];
  const float* W  = (const float*)d_in[1];
  const float* b  = (const float*)d_in[2];
  const float* lA = (const float*)d_in[3];
  const float* lB = (const float*)d_in[4];
  const float* es = (const float*)d_in[5];
  const float* gl = (const float*)d_in[6];
  float* out = (float*)d_out;

  float* Bsc = (float*)d_ws;                   // [OUT_DIM][JR] fp32 (1 MB)
  float* Hws = Bsc + (size_t)OUT_DIM * JR;     // [M_TOK][JR]  fp32 (1 MB)

  sos_lora_prep<<<(OUT_DIM * JR) / 256, 256, 0, stream>>>(lB, es, gl, Bsc);
  sos_lora_h<<<M_TOK / 64, 256, 0, stream>>>(x, lA, Hws);
  dim3 grid(M_TOK / 128, OUT_DIM / 128);
  sos_lora_main<<<grid, 256, 0, stream>>>(x, W, b, Hws, Bsc, out);
}
